// MultiHeadAttention_71674414235938
// MI455X (gfx1250) — compile-verified, matrix operand order checked by reading, not run on hardware
//
#include <hip/hip_runtime.h>

typedef unsigned short u16;
typedef __attribute__((ext_vector_type(8)))  u16    u16x8;
typedef __attribute__((ext_vector_type(16))) u16    u16x16;
typedef __attribute__((ext_vector_type(16))) __bf16 v16bf;
typedef __attribute__((ext_vector_type(8)))  float  v8f;

#define S_LEN 2048
#define HID   1024
#define NHEAD 16
#define DHEAD 64

__device__ __forceinline__ u16 f2bf(float f) {
    unsigned u = __builtin_bit_cast(unsigned, f);
    unsigned r = u + 0x7FFFu + ((u >> 16) & 1u);   // round-to-nearest-even
    return (u16)(r >> 16);
}

__device__ __forceinline__ u16x8 ld8(const u16* p) { return *(const u16x8*)p; }

__device__ __forceinline__ v16bf mkfrag(u16x8 lo, u16x8 hi) {
    u16x16 w = __builtin_shufflevector(lo, hi, 0,1,2,3,4,5,6,7,8,9,10,11,12,13,14,15);
    return __builtin_bit_cast(v16bf, w);
}

__device__ __forceinline__ v8f wmma_bf16(v16bf a, v16bf b, v8f c) {
    return __builtin_amdgcn_wmma_f32_16x16x32_bf16(false, a, false, b, (short)0, c,
                                                   false, false);
}

// CDNA5 async Global->LDS copy (ASYNCcnt-tracked). LDS offset = low 32 bits of
// the generic shared pointer (ISA 10.2).
__device__ __forceinline__ void async_b128(u16* lds_ptr, const u16* gptr) {
    unsigned lds_off = (unsigned)(uintptr_t)lds_ptr;
    asm volatile("global_load_async_to_lds_b128 %0, %1, off"
                 :: "v"(lds_off), "v"(gptr) : "memory");
}
__device__ __forceinline__ void wait_async_le6() {
    asm volatile("s_wait_asynccnt 0x6" ::: "memory");
}
__device__ __forceinline__ void wait_async_0() {
    asm volatile("s_wait_asynccnt 0x0" ::: "memory");
}

// ---------------------------------------------------------------- fp32 -> bf16
__global__ void cvt_bf16(const float* __restrict__ src, u16* __restrict__ dst, int n) {
    int i = blockIdx.x * blockDim.x + threadIdx.x;
    int stride = gridDim.x * blockDim.x;
    for (; i < n; i += stride) dst[i] = f2bf(src[i]);
}

// ------------------------------------------------------------------- GEMM
// C[m,n] = sum_k A[m,k] * W[n,k] + bias[n]
// A: bf16 MxK row-major; W: bf16 NxK row-major (B = W^T).
// mode 0: out bf16 row-major | mode 1: out bf16 transposed (NxM) | mode 2: fp32.
// Block = 8 waves = 128(M) x 64(N) tile. K staged 64-deep through LDS with
// double-buffered async global->LDS copies; every wave reuses the B tile (8x).
// All 10 fragments of a stage are loaded into distinct registers BEFORE the
// 8 WMMAs so the backend can clause the ds_loads and overlap them with the
// matrix pipe instead of serializing on s_wait_dscnt 0.
#define MBLK 128
#define NBLK 64
#define KSTG 64

__global__ __launch_bounds__(256)
void gemm16_bf16(const u16* __restrict__ A, const u16* __restrict__ W,
                 const float* __restrict__ bias, void* __restrict__ out,
                 int M, int N, int K, int mode) {
    __shared__ alignas(16) u16 Abuf[2][MBLK * KSTG];   // 2 x 16 KB
    __shared__ alignas(16) u16 Bbuf[2][NBLK * KSTG];   // 2 x  8 KB

    const int tid   = threadIdx.x;
    const int lane  = tid & 31;
    const int wave  = tid >> 5;
    const int l16   = lane & 15;
    const int off   = (lane < 16) ? 0 : 8;       // K-half select AND row-half select
    const int gm    = blockIdx.x * MBLK;         // block's first A row
    const int gn    = blockIdx.y * NBLK;         // block's first W row (= out col)
    const int ml    = wave * 16 + l16;           // this lane's A row inside tile
    const int mbase = gm + wave * 16;

    v8f acc[4] = {};
    const int nstages = K / KSTG;

    // ---- stage 0 prefetch
    {
#pragma unroll
        for (int i = 0; i < 4; ++i) {            // A tile: 128x64 = 1024 segs
            int seg = tid + i * 256, row = seg >> 3, c8 = (seg & 7) * 8;
            async_b128(&Abuf[0][row * KSTG + c8], A + (size_t)(gm + row) * K + c8);
        }
#pragma unroll
        for (int i = 0; i < 2; ++i) {            // B tile: 64x64 = 512 segs
            int seg = tid + i * 256, row = seg >> 3, c8 = (seg & 7) * 8;
            async_b128(&Bbuf[0][row * KSTG + c8], W + (size_t)(gn + row) * K + c8);
        }
    }

    for (int s = 0; s < nstages; ++s) {
        if (s + 1 < nstages) {                   // prefetch next stage
            const int kb = (s + 1) * KSTG;
            u16* Ad = Abuf[(s + 1) & 1];
            u16* Bd = Bbuf[(s + 1) & 1];
#pragma unroll
            for (int i = 0; i < 4; ++i) {
                int seg = tid + i * 256, row = seg >> 3, c8 = (seg & 7) * 8;
                async_b128(&Ad[row * KSTG + c8], A + (size_t)(gm + row) * K + kb + c8);
            }
#pragma unroll
            for (int i = 0; i < 2; ++i) {
                int seg = tid + i * 256, row = seg >> 3, c8 = (seg & 7) * 8;
                async_b128(&Bd[row * KSTG + c8], W + (size_t)(gn + row) * K + kb + c8);
            }
            wait_async_le6();                    // stage s fully landed (in-order)
        } else {
            wait_async_0();
        }
        __syncthreads();

        const u16* Ab = Abuf[s & 1];
        const u16* Bb = Bbuf[s & 1];

        // load ALL fragments of this stage first (distinct registers)
        const u16* ap = &Ab[ml * KSTG];
        v16bf a0 = mkfrag(ld8(ap + off),      ld8(ap + off + 16));
        v16bf a1 = mkfrag(ld8(ap + 32 + off), ld8(ap + 32 + off + 16));
        v16bf b0[4], b1[4];
#pragma unroll
        for (int t = 0; t < 4; ++t) {
            const u16* bp = &Bb[(t * 16 + l16) * KSTG];
            b0[t] = mkfrag(ld8(bp + off),      ld8(bp + off + 16));
            b1[t] = mkfrag(ld8(bp + 32 + off), ld8(bp + 32 + off + 16));
        }
        // then issue the 8 WMMAs back-to-back
#pragma unroll
        for (int t = 0; t < 4; ++t) acc[t] = wmma_bf16(a0, b0[t], acc[t]);
#pragma unroll
        for (int t = 0; t < 4; ++t) acc[t] = wmma_bf16(a1, b1[t], acc[t]);

        __syncthreads();                          // buffer may be overwritten next
    }

#pragma unroll
    for (int t = 0; t < 4; ++t) {
        const int col = gn + t * 16 + l16;
        const float bv = bias[col];
        if (mode == 0) {
            u16* ob = (u16*)out;
#pragma unroll
            for (int r = 0; r < 8; ++r)
                ob[(size_t)(mbase + r + off) * N + col] = f2bf(acc[t][r] + bv);
        } else if (mode == 1) {
            u16x8 pk = {};
#pragma unroll
            for (int r = 0; r < 8; ++r) pk[r] = f2bf(acc[t][r] + bv);
            *(u16x8*)((u16*)out + (size_t)col * M + mbase + off) = pk;
        } else {
            float* of = (float*)out;
#pragma unroll
            for (int r = 0; r < 8; ++r)
                of[(size_t)(mbase + r + off) * N + col] = acc[t][r] + bv;
        }
    }
}

// ----------------------------------------------------------------- attention
// Flash-style causal attention with modality bias.
// One wave = one (head, 16-query tile). 4 waves/block share a q-tile so
// __syncthreads() trip counts are uniform.
__global__ __launch_bounds__(128)
void attn_flash(const u16* __restrict__ Q, const u16* __restrict__ Kmat,
                const u16* __restrict__ Vt, const float* __restrict__ cmw,
                const long long* __restrict__ mod, u16* __restrict__ ctx) {
    __shared__ alignas(16) u16 lds[4][16 * 32];   // per-wave 16x32 bf16 P tile

    const int lane  = threadIdx.x & 31;
    const int wave  = threadIdx.x >> 5;
    const int l16   = lane & 15;
    const int off   = (lane < 16) ? 0 : 8;
    const int qbase = blockIdx.x * 16;
    const int h     = blockIdx.y * 4 + wave;
    const int hb    = h * DHEAD;
    u16* myl = lds[wave];

    const float c00 = cmw[0], c01 = cmw[1], c02 = cmw[2];
    const float c10 = cmw[3], c11 = cmw[4], c12 = cmw[5];
    const float c20 = cmw[6], c21 = cmw[7], c22 = cmw[8];

    int mq[8];
#pragma unroll
    for (int r = 0; r < 8; ++r) mq[r] = (int)mod[qbase + r + off];

    // Q fragments (d = 0..31 and 32..63), kept in registers for the whole tile
    const u16* qrow = Q + (size_t)(qbase + l16) * HID + hb;
    const v16bf aq0 = mkfrag(ld8(qrow + off),      ld8(qrow + off + 16));
    const v16bf aq1 = mkfrag(ld8(qrow + 32 + off), ld8(qrow + 32 + off + 16));

    v8f acc[4] = {};
    float rmax[8], rsum[8];
#pragma unroll
    for (int r = 0; r < 8; ++r) { rmax[r] = -1e30f; rsum[r] = 0.f; }

    const float kLog2e = 1.44269504f;
    const int nblocks = (qbase + 47) >> 5;          // causal: keys <= qbase+15

    for (int jb = 0; jb < nblocks; ++jb) {
        const int j = jb * 32;

        // ---- scores: two 16x16 tiles over a 32-key block
        const u16* krow1 = Kmat + (size_t)(j + l16) * HID + hb;
        const u16* krow2 = krow1 + (size_t)16 * HID;
        v16bf bk10 = mkfrag(ld8(krow1 + off),      ld8(krow1 + off + 16));
        v16bf bk11 = mkfrag(ld8(krow1 + 32 + off), ld8(krow1 + 32 + off + 16));
        v16bf bk20 = mkfrag(ld8(krow2 + off),      ld8(krow2 + off + 16));
        v16bf bk21 = mkfrag(ld8(krow2 + 32 + off), ld8(krow2 + 32 + off + 16));
        v8f c1 = {}, c2 = {};
        c1 = wmma_bf16(aq0, bk10, c1);
        c1 = wmma_bf16(aq1, bk11, c1);
        c2 = wmma_bf16(aq0, bk20, c2);
        c2 = wmma_bf16(aq1, bk21, c2);

        // ---- modality bias columns for the two key columns this lane owns
        const int mk1 = (int)mod[j + l16];
        const int mk2 = (int)mod[j + 16 + l16];
        const float k1r0 = mk1 == 0 ? c00 : (mk1 == 1 ? c01 : c02);
        const float k1r1 = mk1 == 0 ? c10 : (mk1 == 1 ? c11 : c12);
        const float k1r2 = mk1 == 0 ? c20 : (mk1 == 1 ? c21 : c22);
        const float k2r0 = mk2 == 0 ? c00 : (mk2 == 1 ? c01 : c02);
        const float k2r1 = mk2 == 0 ? c10 : (mk2 == 1 ? c11 : c12);
        const float k2r2 = mk2 == 0 ? c20 : (mk2 == 1 ? c21 : c22);
        const int n1 = j + l16, n2 = j + 16 + l16;

        // ---- online softmax per row (shfl_xor 1/2/4/8 stays inside 16-lane half)
#pragma unroll
        for (int r = 0; r < 8; ++r) {
            const int m = qbase + r + off;
            const float b1 = mq[r] == 0 ? k1r0 : (mq[r] == 1 ? k1r1 : k1r2);
            const float b2 = mq[r] == 0 ? k2r0 : (mq[r] == 1 ? k2r1 : k2r2);
            float s1 = (n1 <= m) ? (c1[r] * 0.125f + b1) : -1e9f;
            float s2 = (n2 <= m) ? (c2[r] * 0.125f + b2) : -1e9f;
            float mx = fmaxf(s1, s2);
            mx = fmaxf(mx, __shfl_xor(mx, 1));
            mx = fmaxf(mx, __shfl_xor(mx, 2));
            mx = fmaxf(mx, __shfl_xor(mx, 4));
            mx = fmaxf(mx, __shfl_xor(mx, 8));
            const float nm   = fmaxf(rmax[r], mx);
            const float p1   = __builtin_amdgcn_exp2f((s1 - nm) * kLog2e);
            const float p2   = __builtin_amdgcn_exp2f((s2 - nm) * kLog2e);
            const float corr = __builtin_amdgcn_exp2f((rmax[r] - nm) * kLog2e);
            rmax[r] = nm;
            float ps = p1 + p2;
            ps += __shfl_xor(ps, 1);
            ps += __shfl_xor(ps, 2);
            ps += __shfl_xor(ps, 4);
            ps += __shfl_xor(ps, 8);
            rsum[r] = rsum[r] * corr + ps;
            acc[0][r] *= corr; acc[1][r] *= corr; acc[2][r] *= corr; acc[3][r] *= corr;
            // scatter P into A-fragment layout (row m, k = column index)
            myl[(r + off) * 32 + l16]      = f2bf(p1);
            myl[(r + off) * 32 + 16 + l16] = f2bf(p2);
        }
        __syncthreads();

        // ---- P (16x32, A-layout) @ V-block (32x64) via Vt
        const v16bf pa = mkfrag(*(const u16x8*)&myl[l16 * 32 + off],
                                *(const u16x8*)&myl[l16 * 32 + off + 16]);
        v16bf bv_[4];
#pragma unroll
        for (int t = 0; t < 4; ++t) {
            const u16* vcol = Vt + (size_t)(hb + t * 16 + l16) * S_LEN + j;
            bv_[t] = mkfrag(ld8(vcol + off), ld8(vcol + off + 16));
        }
#pragma unroll
        for (int t = 0; t < 4; ++t) acc[t] = wmma_bf16(pa, bv_[t], acc[t]);
        __syncthreads();
    }

    // ---- normalize and store ctx (bf16, row-major S x HID)
#pragma unroll
    for (int r = 0; r < 8; ++r) {
        const float inv = 1.0f / rsum[r];
        const size_t rowoff = (size_t)(qbase + r + off) * HID + hb;
#pragma unroll
        for (int t = 0; t < 4; ++t)
            ctx[rowoff + t * 16 + l16] = f2bf(acc[t][r] * inv);
    }
}

// ------------------------------------------------------------------- driver
extern "C" void kernel_launch(void* const* d_in, const int* in_sizes, int n_in,
                              void* d_out, int out_size, void* d_ws, size_t ws_size,
                              hipStream_t stream) {
    const float* x   = (const float*)d_in[0];
    const float* Wq  = (const float*)d_in[1];
    const float* bq  = (const float*)d_in[2];
    const float* Wk  = (const float*)d_in[3];
    const float* bk  = (const float*)d_in[4];
    const float* Wv  = (const float*)d_in[5];
    const float* bv  = (const float*)d_in[6];
    const float* Wo  = (const float*)d_in[7];
    const float* bo  = (const float*)d_in[8];
    const float* cmw = (const float*)d_in[9];
    // d_in[10] = causal mask (structure known analytically -> unused)
    const long long* mod = (const long long*)d_in[11];

    u16* ws = (u16*)d_ws;                 // element (u16) offsets
    const size_t SZ_X = (size_t)S_LEN * HID;   // 2M
    const size_t SZ_W = (size_t)HID * HID;     // 1M
    u16* xb   = ws;
    u16* wqb  = ws + SZ_X;
    u16* wkb  = wqb + SZ_W;
    u16* wvb  = wkb + SZ_W;
    u16* wob  = wvb + SZ_W;
    u16* Qb   = wob + SZ_W;
    u16* Kb   = Qb + SZ_X;
    u16* Vtb  = Kb + SZ_X;
    u16* ctxb = Vtb + SZ_X;

    cvt_bf16<<<1024, 256, 0, stream>>>(x,  xb,  (int)SZ_X);
    cvt_bf16<<<512,  256, 0, stream>>>(Wq, wqb, (int)SZ_W);
    cvt_bf16<<<512,  256, 0, stream>>>(Wk, wkb, (int)SZ_W);
    cvt_bf16<<<512,  256, 0, stream>>>(Wv, wvb, (int)SZ_W);
    cvt_bf16<<<512,  256, 0, stream>>>(Wo, wob, (int)SZ_W);

    dim3 gg(S_LEN / MBLK, HID / NBLK);    // (16, 16), 256-thread blocks
    gemm16_bf16<<<gg, 256, 0, stream>>>(xb, wqb, bq, Qb,  S_LEN, HID, HID, 0);
    gemm16_bf16<<<gg, 256, 0, stream>>>(xb, wkb, bk, Kb,  S_LEN, HID, HID, 0);
    gemm16_bf16<<<gg, 256, 0, stream>>>(xb, wvb, bv, Vtb, S_LEN, HID, HID, 1);

    attn_flash<<<dim3(S_LEN / 16, NHEAD / 4), 128, 0, stream>>>(Qb, Kb, Vtb, cmw,
                                                                mod, ctxb);

    gemm16_bf16<<<gg, 256, 0, stream>>>(ctxb, wob, bo, d_out, S_LEN, HID, HID, 2);
}
